// GaussianKANLayer_70179765617079
// MI455X (gfx1250) — matrix operand order repaired and checked by reading, not compiled, not run on hardware
//
#include <hip/hip_runtime.h>

typedef _Float16 half8 __attribute__((ext_vector_type(8)));
typedef _Float16 v16h  __attribute__((ext_vector_type(16)));
typedef float    v8f   __attribute__((ext_vector_type(8)));

#define NB    16
#define NS    1024
#define DIN   256
#define DOUT  256
#define NG    8
#define NTOT  (NB * NS)      // 16384 samples
#define KDIM  (DIN * NG)     // 2048 reduction length
#define ROWS  32             // sample rows per workgroup
#define LDA   (KDIM + 8)     // padded LDS row stride (halves); keeps 16B align, staggers banks

// ---------------------------------------------------------------------------
// Kernel 0: pack coeffs f32 [DIN, DOUT, G] -> Wp fp16 [DOUT][KDIM] (k-contig),
// and precompute inv_sigma[k] = 1/(exp(log_sigma)+1e-8).
// ---------------------------------------------------------------------------
__global__ __launch_bounds__(256) void kan_pack_kernel(
    const float* __restrict__ coeffs, const float* __restrict__ log_sigma,
    _Float16* __restrict__ wp, float* __restrict__ invsig) {
  int t = blockIdx.x * 256 + threadIdx.x;      // 0 .. DOUT*KDIM-1
  int o = t >> 11;                             // / KDIM
  int k = t & (KDIM - 1);
  int i = k >> 3;
  int g = k & (NG - 1);
  wp[o * KDIM + k] = (_Float16)coeffs[(i * DOUT + o) * NG + g];
  if (t < KDIM) {
    invsig[t] = 1.0f / (__expf(log_sigma[t]) + 1e-8f);
  }
}

// A fragment per ISA 16-bit A 16x32 layout: two 16B reads {k..k+7, k+16..k+23}
__device__ __forceinline__ v16h load_frag_a(const _Float16* p) {
  half8 lo = *(const half8*)(p);
  half8 hi = *(const half8*)(p + 16);
  return __builtin_shufflevector(lo, hi,
           0,1,2,3,4,5,6,7,8,9,10,11,12,13,14,15);
}

// B fragment: two 16B reads {k..k+7, k+16..k+23}, matching A
__device__ __forceinline__ v16h load_frag_b(const _Float16* p) {
  half8 lo = *(const half8*)(p);
  half8 hi = *(const half8*)(p + 16);
  return __builtin_shufflevector(lo, hi,
           0,1,2,3,4,5,6,7,8,9,10,11,12,13,14,15);
}

__device__ __forceinline__ v8f wmma_f16(v16h a, v16h b, v8f c) {
  return __builtin_amdgcn_wmma_f32_16x16x32_f16(
             false, a, false, b, (short)0, c, false, false);
}

// ---------------------------------------------------------------------------
// Kernel 1: per block, build fp16 activation tile A[32][2048] in LDS, then
// 8 waves each compute a 2x2 block of 16x16 output tiles (4 accumulators),
// sharing A/B fragments: 1 ds_load_b128 + 1 global_load_b128 per WMMA.
// ---------------------------------------------------------------------------
__global__ __launch_bounds__(256) void kan_wmma_kernel(
    const float* __restrict__ x, const float* __restrict__ mu,
    const float* __restrict__ invsig, const _Float16* __restrict__ wp,
    float* __restrict__ out) {
  extern __shared__ __align__(16) _Float16 As[];   // ROWS * LDA halves (~129 KB)

  const int tid        = threadIdx.x;
  const int block_row0 = blockIdx.x * ROWS;

  // ---- Stage 1: activations -> LDS (fp16) --------------------------------
  // 32 rows x 256 inputs = 8192 (row,i) pairs; 32 per thread.
  for (int p = tid; p < ROWS * DIN; p += 256) {
    int row = p >> 8;            // / DIN
    int i   = p & (DIN - 1);
    float xt = tanhf(x[(block_row0 + row) * DIN + i]);
    half8 h;
#pragma unroll
    for (int g = 0; g < NG; ++g) {
      float z = (xt - mu[i * NG + g]) * invsig[i * NG + g];
      h[g] = (_Float16)__expf(-z * z);
    }
    *(half8*)(&As[row * LDA + i * NG]) = h;   // 16B aligned store
  }
  __syncthreads();

  // ---- Stage 2: WMMA GEMM, 2x2 register blocking per wave ----------------
  const int wave = tid >> 5;      // 0..7 -> col-tile pair
  const int lane = tid & 31;
  const int m    = lane & 15;     // A row / B col / C col within tile
  const int hi   = lane >> 4;     // half-wave select
  const int ka   = hi * 8;        // A K sub-offset (halves): {0-7,16-23} vs {8-15,24-31}
  const int kb   = hi * 8;        // B K sub-offset (halves): {0-7,16-23} vs {8-15,24-31}

  const int ct0 = wave * 2;       // this wave's two col-tiles: ct0, ct0+1

  v8f acc00 = {}, acc01 = {}, acc10 = {}, acc11 = {};   // [row-tile][col-tile]

  const _Float16* __restrict__ arow0 = &As[(0 * 16 + m) * LDA] + ka;
  const _Float16* __restrict__ arow1 = &As[(1 * 16 + m) * LDA] + ka;
  const _Float16* __restrict__ bcol0 = wp + (size_t)((ct0 + 0) * 16 + m) * KDIM + kb;
  const _Float16* __restrict__ bcol1 = wp + (size_t)((ct0 + 1) * 16 + m) * KDIM + kb;

  for (int kt = 0; kt < KDIM / 32; ++kt) {
    int k0 = kt * 32;
    v16h a0 = load_frag_a(arow0 + k0);
    v16h a1 = load_frag_a(arow1 + k0);
    v16h b0 = load_frag_b(bcol0 + k0);
    v16h b1 = load_frag_b(bcol1 + k0);
    acc00 = wmma_f16(a0, b0, acc00);
    acc01 = wmma_f16(a0, b1, acc01);
    acc10 = wmma_f16(a1, b0, acc10);
    acc11 = wmma_f16(a1, b1, acc11);
  }

  // ---- Store: C layout VGPR v -> M = v + hi*8, N = lane&15 ---------------
  const int col0 = ct0 * 16 + m;
#pragma unroll
  for (int v = 0; v < 8; ++v) {
    int n0 = block_row0 + 0 * 16 + hi * 8 + v;
    int n1 = block_row0 + 1 * 16 + hi * 8 + v;
    out[(size_t)n0 * DOUT + col0]      = acc00[v];
    out[(size_t)n0 * DOUT + col0 + 16] = acc01[v];
    out[(size_t)n1 * DOUT + col0]      = acc10[v];
    out[(size_t)n1 * DOUT + col0 + 16] = acc11[v];
  }
}

// ---------------------------------------------------------------------------
extern "C" void kernel_launch(void* const* d_in, const int* in_sizes, int n_in,
                              void* d_out, int out_size, void* d_ws, size_t ws_size,
                              hipStream_t stream) {
  const float* x         = (const float*)d_in[0];   // [16,1024,256]
  const float* mu        = (const float*)d_in[1];   // [256,8]
  const float* log_sigma = (const float*)d_in[2];   // [256,8]
  const float* coeffs    = (const float*)d_in[3];   // [256,256,8]
  float* out = (float*)d_out;

  // workspace layout: [0, 1MB) Wp fp16 [DOUT][KDIM]; then invsig f32 [KDIM]
  _Float16* wp   = (_Float16*)d_ws;
  float* invsig  = (float*)((char*)d_ws + (size_t)DOUT * KDIM * sizeof(_Float16));

  kan_pack_kernel<<<(DOUT * KDIM) / 256, 256, 0, stream>>>(coeffs, log_sigma, wp, invsig);

  size_t lds_bytes = (size_t)ROWS * LDA * sizeof(_Float16);
  kan_wmma_kernel<<<NTOT / ROWS, 256, lds_bytes, stream>>>(x, mu, invsig, wp, out);
}
